// MultiHeadAttention_30331059044419
// MI455X (gfx1250) — matrix operand order repaired and checked by reading, not compiled, not run on hardware
//
#include <hip/hip_runtime.h>
#include <math.h>

// ---------------- CDNA5 WMMA types ----------------
typedef __attribute__((ext_vector_type(16))) __bf16 v16bf;
typedef __attribute__((ext_vector_type(8)))  __bf16 v8bf;
typedef __attribute__((ext_vector_type(8)))  float  v8f;

#define DEV __device__ __forceinline__

constexpr int   kB     = 2;
constexpr int   kN     = 2048;   // Nq == Nk
constexpr int   kD     = 1024;   // d_model
constexpr int   kH     = 16;     // heads
constexpr int   kDh    = 64;     // head dim
constexpr float kScale = 0.125f; // 64^-0.5

DEV v8f wmma_bf16(v16bf a, v16bf b, v8f c) {
  return __builtin_amdgcn_wmma_f32_16x16x32_bf16(false, a, false, b,
                                                 (short)0, c, false, false);
}

// ---------------- fp32 -> bf16 conversion ----------------
__global__ void cvt_bf16_kernel(const float* __restrict__ src,
                                __bf16* __restrict__ dst, int n) {
  int i = blockIdx.x * blockDim.x + threadIdx.x;
  if (i < n) dst[i] = (__bf16)src[i];
}

// A-fragment for v_wmma_f32_16x16x32_bf16 (16x32, M x K):
//  lane L (L<16):  row M=L,    K in {k0+0..7,  k0+16..23}
//  lane L (L>=16): row M=L-16, K in {k0+8..15, k0+24..31}
DEV v16bf load_a_frag(const __bf16* __restrict__ rowptr, int k0, int hi) {
  union { v16bf v; v8bf h[2]; } u;
  u.h[0] = *reinterpret_cast<const v8bf*>(rowptr + k0 + 8 * hi);
  u.h[1] = *reinterpret_cast<const v8bf*>(rowptr + k0 + 16 + 8 * hi);
  return u.v;
}

// B-fragment (32x16, K x N): lane holds col N=lane%16, K in {k0+8*hi+0..7, k0+16+8*hi+0..7}
// -> two 16-byte loads when memory is [N][K] row-major (matches A-fragment K order).
DEV v16bf load_b_frag(const __bf16* __restrict__ colrow, int k0, int hi) {
  union { v16bf v; v8bf h[2]; } u;
  u.h[0] = *reinterpret_cast<const v8bf*>(colrow + k0 + 8 * hi);
  u.h[1] = *reinterpret_cast<const v8bf*>(colrow + k0 + 16 + 8 * hi);
  return u.v;
}

// ---------------- tiled bf16 WMMA GEMM:  C[M,N] = A[M,K] @ B[N,K]^T ----------------
// Wave tile: 32 rows x 64 cols (2 A-row fragments reuse 4 B fragments -> 8 WMMA/K-step).
// Block: 8 waves stacked in M -> 256 x 64 block tile.
// out_mode 0: bf16 out, layout [b][h][tok][d]   (row=b*kN+tok, col=h*64+d)
// out_mode 1: bf16 out, layout [b][h][d][tok]   (transposed V)
// out_mode 2: f32  out, out[row*N+col] = acc + bias[col]
__global__ void __launch_bounds__(256)
gemm_bf16_kernel(const __bf16* __restrict__ A, const __bf16* __restrict__ Bm,
                 const float* __restrict__ bias, void* __restrict__ out,
                 int M, int N, int K, int out_mode) {
  const int wave = threadIdx.x >> 5;
  const int lane = threadIdx.x & 31;
  const int hi   = lane >> 4;
  const int lm   = lane & 15;

  const int row0 = blockIdx.y * 256 + wave * 32;
  const int col0 = blockIdx.x * 64;

  const __bf16* arow0 = A + (size_t)(row0 + lm) * K;
  const __bf16* arow1 = A + (size_t)(row0 + 16 + lm) * K;
  const __bf16* brow  = Bm + (size_t)(col0 + lm) * K;   // t-th tile adds 16*K

  v8f acc[2][4] = {};

  for (int k0 = 0; k0 < K; k0 += 32) {
    v16bf a0 = load_a_frag(arow0, k0, hi);
    v16bf a1 = load_a_frag(arow1, k0, hi);
#pragma unroll
    for (int t = 0; t < 4; ++t) {
      v16bf b = load_b_frag(brow + (size_t)(16 * t) * K, k0, hi);
      acc[0][t] = wmma_bf16(a0, b, acc[0][t]);
      acc[1][t] = wmma_bf16(a1, b, acc[1][t]);
    }
  }

  // C/D layout: VGPR r holds M = 16*mt + r + 8*hi, N = lm
#pragma unroll
  for (int mt = 0; mt < 2; ++mt) {
#pragma unroll
    for (int t = 0; t < 4; ++t) {
#pragma unroll
      for (int r = 0; r < 8; ++r) {
        const int m = row0 + 16 * mt + r + 8 * hi;
        const int n = col0 + 16 * t + lm;
        const float v = acc[mt][t][r];
        if (out_mode == 2) {
          reinterpret_cast<float*>(out)[(size_t)m * N + n] = v + bias[n];
        } else {
          const int bb = m >> 11, tok = m & (kN - 1);   // m = b*2048 + tok
          const int h  = n >> 6,  d   = n & 63;         // n = h*64 + d
          __bf16* o = reinterpret_cast<__bf16*>(out);
          if (out_mode == 0)
            o[(((size_t)(bb * kH + h) * kN + tok) * kDh) + d] = (__bf16)v;
          else  // transposed V: [b][h][d][tok]
            o[(((size_t)(bb * kH + h) * kDh + d) * kN) + tok] = (__bf16)v;
        }
      }
    }
  }
}

// ---------------- flash attention: O = softmax(Q K^T * scale) V ----------------
// Qh,Kh: [B,H,N,64] bf16 ; Vt: [B,H,64,N] bf16 ; att out: [B,N, H*64] bf16
// Wave handles 32 query rows (2 M-tiles) so each K/V fragment feeds two WMMAs.
__global__ void __launch_bounds__(256)
attn_kernel(const __bf16* __restrict__ Qh, const __bf16* __restrict__ Kh,
            const __bf16* __restrict__ Vt, __bf16* __restrict__ att) {
  const int wave = threadIdx.x >> 5;
  const int lane = threadIdx.x & 31;
  const int hi   = lane >> 4;
  const int lm   = lane & 15;

  const int h  = blockIdx.y;
  const int b  = blockIdx.z;
  const int q0 = blockIdx.x * 256 + wave * 32;   // 32 query rows per wave

  const size_t headQK = (size_t)(b * kH + h) * kN * kDh;
  const __bf16* qrow0 = Qh + headQK + (size_t)(q0 + lm) * kDh;
  const __bf16* qrow1 = Qh + headQK + (size_t)(q0 + 16 + lm) * kDh;
  const __bf16* kbase = Kh + headQK;
  const __bf16* vbase = Vt + (size_t)(b * kH + h) * kDh * kN;

  // Q fragments reused across the whole key loop (Dh = 64 = 2 K-tiles, 2 M-tiles)
  v16bf qfrag[2][2];
#pragma unroll
  for (int kt = 0; kt < 2; ++kt) {
    qfrag[0][kt] = load_a_frag(qrow0, 32 * kt, hi);
    qfrag[1][kt] = load_a_frag(qrow1, 32 * kt, hi);
  }

  v8f o[2][4] = {};
  float rmax[2][8], rsum[2][8];
#pragma unroll
  for (int mt = 0; mt < 2; ++mt)
#pragma unroll
    for (int r = 0; r < 8; ++r) {
      rmax[mt][r] = -__builtin_huge_valf();
      rsum[mt][r] = 0.f;
    }

  // per-wave LDS tile: P is 32x32 bf16, rows padded to 40 (80B, 16B aligned)
  __shared__ __bf16 plds[8][32][40];

  for (int j0 = 0; j0 < kN; j0 += 32) {
    // ---- S = (Q K^T): 2 M-tiles x 2 key-tiles x 2 Dh-tiles = 8 WMMA, 4 K-frag loads
    v8f s[2][2] = {};
#pragma unroll
    for (int nt = 0; nt < 2; ++nt) {
      const __bf16* krow = kbase + (size_t)(j0 + 16 * nt + lm) * kDh;
#pragma unroll
      for (int kt = 0; kt < 2; ++kt) {
        v16bf kf = load_b_frag(krow, 32 * kt, hi);
        s[0][nt] = wmma_bf16(qfrag[0][kt], kf, s[0][nt]);
        s[1][nt] = wmma_bf16(qfrag[1][kt], kf, s[1][nt]);
      }
    }

    // ---- online softmax (row M = 16*mt + r + 8*hi lives on 16 lanes sharing hi)
#pragma unroll
    for (int mt = 0; mt < 2; ++mt) {
#pragma unroll
      for (int r = 0; r < 8; ++r) {
        float s0 = s[mt][0][r] * kScale;
        float s1 = s[mt][1][r] * kScale;
        float mloc = fmaxf(s0, s1);
#pragma unroll
        for (int d = 1; d < 16; d <<= 1)
          mloc = fmaxf(mloc, __shfl_xor(mloc, d, 32));
        const float nm    = fmaxf(rmax[mt][r], mloc);
        const float alpha = __expf(rmax[mt][r] - nm);
        rmax[mt][r] = nm;
        const float p0 = __expf(s0 - nm);
        const float p1 = __expf(s1 - nm);
        s[mt][0][r] = p0; s[mt][1][r] = p1;
        float ps = p0 + p1;
#pragma unroll
        for (int d = 1; d < 16; d <<= 1)
          ps += __shfl_xor(ps, d, 32);
        rsum[mt][r] = rsum[mt][r] * alpha + ps;
#pragma unroll
        for (int t = 0; t < 4; ++t) o[mt][t][r] *= alpha;
      }
    }

    // ---- P: C-layout -> LDS -> A-fragment layout
#pragma unroll
    for (int mt = 0; mt < 2; ++mt)
#pragma unroll
      for (int r = 0; r < 8; ++r) {
        const int m = 16 * mt + r + 8 * hi;
        plds[wave][m][lm]      = (__bf16)s[mt][0][r];
        plds[wave][m][16 + lm] = (__bf16)s[mt][1][r];
      }
    __syncthreads();   // uniform across the block; orders LDS store->load

    v16bf pfrag[2];
    pfrag[0] = load_a_frag(&plds[wave][lm][0], 0, hi);
    pfrag[1] = load_a_frag(&plds[wave][16 + lm][0], 0, hi);

    // ---- O += P (32x32) @ V (32x64): 8 WMMA, 4 V-frag loads (Vt layout)
#pragma unroll
    for (int t = 0; t < 4; ++t) {
      const __bf16* vrow = vbase + (size_t)(16 * t + lm) * kN + j0;
      v16bf vf = load_b_frag(vrow, 0, hi);
      o[0][t] = wmma_bf16(pfrag[0], vf, o[0][t]);
      o[1][t] = wmma_bf16(pfrag[1], vf, o[1][t]);
    }
    __syncthreads();   // protect LDS tile before next iteration's stores
  }

  // ---- normalize and write att[b][q][h*64+d] (bf16, consumed by Wo GEMM)
#pragma unroll
  for (int mt = 0; mt < 2; ++mt)
#pragma unroll
    for (int r = 0; r < 8; ++r) {
      const float inv = 1.0f / rsum[mt][r];
      const int   q   = q0 + 16 * mt + r + 8 * hi;
#pragma unroll
      for (int t = 0; t < 4; ++t) {
        const int d = 16 * t + lm;
        att[((size_t)(b * kN + q) * kD) + h * kDh + d] =
            (__bf16)(o[mt][t][r] * inv);
      }
    }
}

// ---------------- host-side orchestration ----------------
extern "C" void kernel_launch(void* const* d_in, const int* in_sizes, int n_in,
                              void* d_out, int out_size, void* d_ws, size_t ws_size,
                              hipStream_t stream) {
  (void)in_sizes; (void)n_in; (void)out_size; (void)ws_size;

  const float* f_query = (const float*)d_in[0];
  const float* f_key   = (const float*)d_in[1];
  const float* f_value = (const float*)d_in[2];
  const float* f_Wq    = (const float*)d_in[3];
  const float* f_Wk    = (const float*)d_in[4];
  const float* f_Wv    = (const float*)d_in[5];
  const float* f_Wo    = (const float*)d_in[6];
  const float* f_bo    = (const float*)d_in[7];

  const size_t nX = (size_t)kB * kN * kD;  // 4,194,304
  const size_t nW = (size_t)kD * kD;       // 1,048,576

  __bf16* ws  = reinterpret_cast<__bf16*>(d_ws);
  __bf16* xq  = ws;            // [B*N, D]
  __bf16* xk  = xq  + nX;
  __bf16* xv  = xk  + nX;
  __bf16* wq  = xv  + nX;      // [D, D]
  __bf16* wk  = wq  + nW;
  __bf16* wv  = wk  + nW;
  __bf16* wo  = wv  + nW;
  __bf16* qh  = wo  + nW;      // [B,H,N,64]
  __bf16* kh  = qh  + nX;      // [B,H,N,64]
  __bf16* vt  = kh  + nX;      // [B,H,64,N]
  __bf16* att = vt  + nX;      // [B,N,D]
  // total: 7*nX + 4*nW bf16 = 64 MB of d_ws

  const int cvtThreads = 256;
  auto cvt = [&](const float* s, __bf16* d, size_t n) {
    cvt_bf16_kernel<<<dim3((unsigned)((n + cvtThreads - 1) / cvtThreads)),
                      cvtThreads, 0, stream>>>(s, d, (int)n);
  };
  cvt(f_query, xq, nX);
  cvt(f_key,   xk, nX);
  cvt(f_value, xv, nX);
  cvt(f_Wq,    wq, nW);
  cvt(f_Wk,    wk, nW);
  cvt(f_Wv,    wv, nW);
  cvt(f_Wo,    wo, nW);

  const int M = kB * kN;                   // 4096
  const dim3 ggrid(kD / 64, M / 256);      // 16 x 16
  const dim3 gblk(256);

  // projections -> attention layouts
  gemm_bf16_kernel<<<ggrid, gblk, 0, stream>>>(xq, wq, nullptr, qh, M, kD, kD, 0);
  gemm_bf16_kernel<<<ggrid, gblk, 0, stream>>>(xk, wk, nullptr, kh, M, kD, kD, 0);
  gemm_bf16_kernel<<<ggrid, gblk, 0, stream>>>(xv, wv, nullptr, vt, M, kD, kD, 1);

  // flash attention per (b, h, 256-query tile)
  attn_kernel<<<dim3(kN / 256, kH, kB), gblk, 0, stream>>>(qh, kh, vt, att);

  // output projection + bias -> f32 d_out
  gemm_bf16_kernel<<<ggrid, gblk, 0, stream>>>(att, wo, f_bo, d_out, M, kD, kD, 2);
}
